// LearnedSimulator_68539088110029
// MI455X (gfx1250) — matrix operand order repaired and checked by reading, not compiled, not run on hardware
//
#include <hip/hip_runtime.h>

#define Hdim 128

typedef __attribute__((ext_vector_type(16))) _Float16 v16h;
typedef __attribute__((ext_vector_type(4)))  _Float16 h4;
typedef __attribute__((ext_vector_type(8)))  float    v8f;
typedef __attribute__((ext_vector_type(4)))  float    f4;

union ABFrag { v16h h; f4 q[2]; };

__device__ __forceinline__ h4 cvt_h4(f4 v) {
  h4 o;
  o.x = (_Float16)v.x; o.y = (_Float16)v.y;
  o.z = (_Float16)v.z; o.w = (_Float16)v.w;
  return o;
}

// A fragment (16x32 f16): lane 0-15 = row M, holds K=kt*32+{0..7,16..23};
// lanes 16-31 hold K=kt*32+{8..15,24..31}. Two 16B LDS loads.
__device__ __forceinline__ v16h load_afrag(const _Float16* X, int XSH, int kt, int lane) {
  int row = lane & 15;
  int hi  = lane >> 4;
  const _Float16* p = X + row * XSH + kt * 32 + hi * 8;
  ABFrag a;
  a.q[0] = *(const f4*)(p);
  a.q[1] = *(const f4*)(p + 16);
  return a.h;
}

// B fragment: pre-swizzled so each lane's 16 halves are contiguous (32B).
__device__ __forceinline__ v16h load_bfrag(const _Float16* W, int kt, int nt, int lane) {
  const _Float16* p = W + (((kt << 3) + nt) * 32 + lane) * 16;
  ABFrag b;
  b.q[0] = *(const f4*)(p);
  b.q[1] = *(const f4*)(p + 8);
  return b.h;
}

// Chained GEMM sub-layers on a 16-row tile in per-wave LDS, with A/B fragment
// double-buffering so ds_load latency overlaps the WMMA pipe.
// Intermediate layers: +bias, relu, f16 back in place. Last: +bias f32 in place.
template<int NSUB, bool RELU_LAST>
__device__ __forceinline__ void mlp_chain(_Float16* X, int XSH, const _Float16* W, int K0,
                                          const float* biasL, int lane) {
  int woff = 0;
  #pragma unroll
  for (int li = 0; li < NSUB; ++li) {
    const int K = (li == 0) ? K0 : Hdim;
    const _Float16* Wl = W + woff;
    woff += K * Hdim;
    v8f zero = {0.f,0.f,0.f,0.f,0.f,0.f,0.f,0.f};
    v8f acc[8];
    #pragma unroll
    for (int nt = 0; nt < 8; ++nt) acc[nt] = zero;
    const int KT = K >> 5;
    v16h a = load_afrag(X, XSH, 0, lane);
    v16h b = load_bfrag(Wl, 0, 0, lane);
    for (int kt = 0; kt < KT; ++kt) {
      int ktn = (kt + 1 < KT) ? kt + 1 : kt;
      v16h a_next = load_afrag(X, XSH, ktn, lane);
      #pragma unroll
      for (int nt = 0; nt < 8; ++nt) {
        v16h b_next = (nt < 7) ? load_bfrag(Wl, kt, nt + 1, lane)
                               : load_bfrag(Wl, ktn, 0, lane);
        acc[nt] = __builtin_amdgcn_wmma_f32_16x16x32_f16(false, a, false, b,
                                                         (short)0, acc[nt], false, false);
        b = b_next;
      }
      a = a_next;
    }
    const int col0  = lane & 15;
    const int mbase = (lane >> 4) * 8;
    const float* bias = biasL + li * Hdim;
    if (li < NSUB - 1) {
      #pragma unroll
      for (int nt = 0; nt < 8; ++nt) {
        int col = (nt << 4) + col0;
        float bc = bias[col];
        #pragma unroll
        for (int v = 0; v < 8; ++v) {
          float val = acc[nt][v] + bc;
          val = fmaxf(val, 0.f);
          X[(mbase + v) * XSH + col] = (_Float16)val;
        }
      }
    } else {
      float* Xf = (float*)X;
      const int XSF = XSH >> 1;
      #pragma unroll
      for (int nt = 0; nt < 8; ++nt) {
        int col = (nt << 4) + col0;
        float bc = bias[col];
        #pragma unroll
        for (int v = 0; v < 8; ++v) {
          float val = acc[nt][v] + bc;
          if (RELU_LAST) val = fmaxf(val, 0.f);
          Xf[(mbase + v) * XSF + col] = val;
        }
      }
    }
  }
}

__device__ __forceinline__ void row_stats(const float* Xf, int XSF, float* stats, int lane) {
  if (lane < 16) {
    const f4* row = (const f4*)(Xf + lane * XSF);
    float s1 = 0.f, s2 = 0.f;
    for (int c = 0; c < 32; ++c) {
      f4 v = row[c];
      s1 += v.x + v.y + v.z + v.w;
      s2 += v.x * v.x + v.y * v.y + v.z * v.z + v.w * v.w;
    }
    float mu  = s1 * (1.f / 128.f);
    float var = s2 * (1.f / 128.f) - mu * mu;
    if (var < 0.f) var = 0.f;
    stats[lane * 2 + 0] = mu;
    stats[lane * 2 + 1] = rsqrtf(var + 1e-5f);
  }
}

// ---- weight pre-swizzle: fp32 KxN(row-major, N=128) -> f16 B-fragment order ----
__global__ void k_swizzle(const float* __restrict__ src, _Float16* __restrict__ dst,
                          int K, int Kpad, int batch, long long dstStride) {
  long long per = (long long)Kpad * 128;
  long long total = per * batch;
  for (long long idx = blockIdx.x * (long long)blockDim.x + threadIdx.x; idx < total;
       idx += (long long)gridDim.x * blockDim.x) {
    long long b = idx / per;
    long long r = idx - b * per;
    int j    = (int)(r & 15);
    int lane = (int)((r >> 4) & 31);
    long long tile = r >> 9;
    int nt = (int)(tile & 7);
    int kt = (int)(tile >> 3);
    int n = (nt << 4) + (lane & 15);
    int k = (kt << 5) + ((j >> 3) << 4) + ((lane >> 4) << 3) + (j & 7);
    float v = (k < K) ? src[b * (long long)K * 128 + (long long)k * 128 + n] : 0.f;
    dst[b * dstStride + r] = (_Float16)v;
  }
}

__global__ void k_zero(float* p, long long nElem) {
  for (long long i = blockIdx.x * (long long)blockDim.x + threadIdx.x; i < nElem;
       i += (long long)gridDim.x * blockDim.x)
    p[i] = 0.f;
}

// ---- node encoder: [emb[x] | pos | pad] (32) -> 128, LN ----
__global__ void __launch_bounds__(256, 1)
k_node_enc(const int* __restrict__ xt, const float* __restrict__ pos,
           const float* __restrict__ emb, float* __restrict__ nf, int N, int nTiles,
           const _Float16* __restrict__ Wsw,
           const float* b0, const float* b1, const float* b2, const float* b3,
           const float* be, const float* g) {
  extern __shared__ char lds[];
  const int WH = 32 * 128 + 3 * 128 * 128;   // 53248 halves
  const int XSH = 280;
  _Float16* Wb    = (_Float16*)lds;
  _Float16* Xall  = (_Float16*)(lds + WH * 2);
  float* statsAll = (float*)(lds + WH * 2 + 8 * 16 * XSH * 2);
  float* pL       = statsAll + 8 * 32;
  int tid = threadIdx.x, lane = tid & 31, wave = tid >> 5;
  _Float16* X = Xall + wave * 16 * XSH;
  float* stats = statsAll + wave * 32;
  for (int i = tid; i < WH / 8; i += blockDim.x) ((f4*)Wb)[i] = ((const f4*)Wsw)[i];
  for (int i = tid; i < 128; i += blockDim.x) {
    pL[i] = b0[i]; pL[128 + i] = b1[i]; pL[256 + i] = b2[i]; pL[384 + i] = b3[i];
    pL[512 + i] = g[i]; pL[640 + i] = be[i];
  }
  __syncthreads();
  f4 gq = *(const f4*)(pL + 512 + 4 * lane);
  f4 bq = *(const f4*)(pL + 640 + 4 * lane);
  for (int tg = blockIdx.x; tg < nTiles; tg += gridDim.x) {
    int n0 = tg * 128 + wave * 16;
    for (int r = 0; r < 16; ++r) {
      int n = n0 + r; bool ok = n < N;
      int c = lane;
      float v = 0.f;
      if (ok) {
        if (c < 16)      v = emb[xt[n] * 16 + c];
        else if (c < 22) v = pos[(size_t)n * 6 + (c - 16)];
      }
      X[r * XSH + c] = (_Float16)v;
    }
    mlp_chain<4, false>(X, XSH, Wb, 32, pL, lane);
    float* Xf = (float*)X; const int XSF = XSH >> 1;
    row_stats(Xf, XSF, stats, lane);
    __syncthreads();
    for (int r = 0; r < 16; ++r) {
      int n = n0 + r; if (n >= N) break;
      float mu = stats[r * 2], rs = stats[r * 2 + 1];
      f4 xv = *(const f4*)(Xf + r * XSF + 4 * lane);
      f4 v;
      v.x = (xv.x - mu) * rs * gq.x + bq.x;
      v.y = (xv.y - mu) * rs * gq.y + bq.y;
      v.z = (xv.z - mu) * rs * gq.z + bq.z;
      v.w = (xv.w - mu) * rs * gq.w + bq.w;
      ((f4*)(nf + (size_t)n * 128))[lane] = v;
    }
    __syncthreads();
  }
}

// ---- edge encoder: [edge_attr | pad] (32) -> 128, LN ----
__global__ void __launch_bounds__(256, 1)
k_edge_enc(const float* __restrict__ eattr, float* __restrict__ ef, int E, int nTiles,
           const _Float16* __restrict__ Wsw,
           const float* b0, const float* b1, const float* b2, const float* b3,
           const float* be, const float* g) {
  extern __shared__ char lds[];
  const int WH = 32 * 128 + 3 * 128 * 128;
  const int XSH = 280;
  _Float16* Wb    = (_Float16*)lds;
  _Float16* Xall  = (_Float16*)(lds + WH * 2);
  float* statsAll = (float*)(lds + WH * 2 + 8 * 16 * XSH * 2);
  float* pL       = statsAll + 8 * 32;
  int tid = threadIdx.x, lane = tid & 31, wave = tid >> 5;
  _Float16* X = Xall + wave * 16 * XSH;
  float* stats = statsAll + wave * 32;
  for (int i = tid; i < WH / 8; i += blockDim.x) ((f4*)Wb)[i] = ((const f4*)Wsw)[i];
  for (int i = tid; i < 128; i += blockDim.x) {
    pL[i] = b0[i]; pL[128 + i] = b1[i]; pL[256 + i] = b2[i]; pL[384 + i] = b3[i];
    pL[512 + i] = g[i]; pL[640 + i] = be[i];
  }
  __syncthreads();
  f4 gq = *(const f4*)(pL + 512 + 4 * lane);
  f4 bq = *(const f4*)(pL + 640 + 4 * lane);
  for (int tg = blockIdx.x; tg < nTiles; tg += gridDim.x) {
    int e0 = tg * 128 + wave * 16;
    for (int r = 0; r < 16; ++r) {
      int e = e0 + r; bool ok = e < E;
      int c = lane;
      float v = (ok && c < 3) ? eattr[(size_t)e * 3 + c] : 0.f;
      X[r * XSH + c] = (_Float16)v;
    }
    mlp_chain<4, false>(X, XSH, Wb, 32, pL, lane);
    float* Xf = (float*)X; const int XSF = XSH >> 1;
    row_stats(Xf, XSF, stats, lane);
    __syncthreads();
    for (int r = 0; r < 16; ++r) {
      int e = e0 + r; if (e >= E) break;
      float mu = stats[r * 2], rs = stats[r * 2 + 1];
      f4 xv = *(const f4*)(Xf + r * XSF + 4 * lane);
      f4 v;
      v.x = (xv.x - mu) * rs * gq.x + bq.x;
      v.y = (xv.y - mu) * rs * gq.y + bq.y;
      v.z = (xv.z - mu) * rs * gq.z + bq.z;
      v.w = (xv.w - mu) * rs * gq.w + bq.w;
      ((f4*)(ef + (size_t)e * 128))[lane] = v;
    }
    __syncthreads();
  }
}

// ---- edge message layer: [nf[dst]|nf[src]|ef] (384) -> 128, LN, ef+=, atomic aggr ----
__global__ void __launch_bounds__(256, 1)
k_edge_layer(const float* __restrict__ nf, float* __restrict__ ef, float* __restrict__ aggr,
             const int* __restrict__ eidx, int E, int nTiles,
             const _Float16* __restrict__ Wsw,
             const float* b0, const float* b1, const float* b2, const float* b3,
             const float* be, const float* g) {
  extern __shared__ char lds[];
  const int WH = 384 * 128 + 3 * 128 * 128;   // 98304 halves -> 196.6KB
  const int XSH = 408;
  _Float16* Wb    = (_Float16*)lds;
  _Float16* Xall  = (_Float16*)(lds + WH * 2);
  float* statsAll = (float*)(lds + WH * 2 + 8 * 16 * XSH * 2);
  float* pL       = statsAll + 8 * 32;
  int tid = threadIdx.x, lane = tid & 31, wave = tid >> 5;
  _Float16* X = Xall + wave * 16 * XSH;
  float* stats = statsAll + wave * 32;
  for (int i = tid; i < WH / 8; i += blockDim.x) ((f4*)Wb)[i] = ((const f4*)Wsw)[i];
  for (int i = tid; i < 128; i += blockDim.x) {
    pL[i] = b0[i]; pL[128 + i] = b1[i]; pL[256 + i] = b2[i]; pL[384 + i] = b3[i];
    pL[512 + i] = g[i]; pL[640 + i] = be[i];
  }
  __syncthreads();
  const int* srcIdx = eidx;
  const int* dstIdx = eidx + E;
  const f4 z4 = {0.f,0.f,0.f,0.f};
  f4 gq = *(const f4*)(pL + 512 + 4 * lane);
  f4 bq = *(const f4*)(pL + 640 + 4 * lane);
  for (int tg = blockIdx.x; tg < nTiles; tg += gridDim.x) {
    int e0 = tg * 128 + wave * 16;
    for (int r = 0; r < 16; ++r) {
      int e = e0 + r;
      bool ok = (e < E);
      int d = ok ? dstIdx[e] : 0;
      int s = ok ? srcIdx[e] : 0;
      f4 vd = ok ? ((const f4*)(nf + (size_t)d * 128))[lane] : z4;
      f4 vs = ok ? ((const f4*)(nf + (size_t)s * 128))[lane] : z4;
      f4 ve = ok ? ((const f4*)(ef + (size_t)e * 128))[lane] : z4;
      h4* xr = (h4*)(X + r * XSH);
      xr[lane]      = cvt_h4(vd);
      xr[32 + lane] = cvt_h4(vs);
      xr[64 + lane] = cvt_h4(ve);
    }
    mlp_chain<4, false>(X, XSH, Wb, 384, pL, lane);
    float* Xf = (float*)X; const int XSF = XSH >> 1;
    row_stats(Xf, XSF, stats, lane);
    __syncthreads();
    for (int r = 0; r < 16; ++r) {
      int e = e0 + r; if (e >= E) break;
      float mu = stats[r * 2], rs = stats[r * 2 + 1];
      int d = dstIdx[e];
      f4 xv = *(const f4*)(Xf + r * XSF + 4 * lane);
      f4 v;
      v.x = (xv.x - mu) * rs * gq.x + bq.x;
      v.y = (xv.y - mu) * rs * gq.y + bq.y;
      v.z = (xv.z - mu) * rs * gq.z + bq.z;
      v.w = (xv.w - mu) * rs * gq.w + bq.w;
      f4* pe = (f4*)(ef + (size_t)e * 128);
      f4 o = pe[lane];
      o.x += v.x; o.y += v.y; o.z += v.z; o.w += v.w;
      pe[lane] = o;
      float* ap = aggr + (size_t)d * 128 + 4 * lane;
      atomicAdd(ap + 0, v.x); atomicAdd(ap + 1, v.y);
      atomicAdd(ap + 2, v.z); atomicAdd(ap + 3, v.w);
    }
    __syncthreads();
  }
}

// ---- node update layer: [nf|aggr] (256) -> 128, LN, nf+=; rezero aggr ----
__global__ void __launch_bounds__(256, 1)
k_node_layer(float* __restrict__ nf, float* __restrict__ aggr, int N, int nTiles,
             const _Float16* __restrict__ Wsw,
             const float* b0, const float* b1, const float* b2, const float* b3,
             const float* be, const float* g) {
  extern __shared__ char lds[];
  const int WH = 256 * 128 + 3 * 128 * 128;   // 81920 halves
  const int XSH = 280;
  _Float16* Wb    = (_Float16*)lds;
  _Float16* Xall  = (_Float16*)(lds + WH * 2);
  float* statsAll = (float*)(lds + WH * 2 + 8 * 16 * XSH * 2);
  float* pL       = statsAll + 8 * 32;
  int tid = threadIdx.x, lane = tid & 31, wave = tid >> 5;
  _Float16* X = Xall + wave * 16 * XSH;
  float* stats = statsAll + wave * 32;
  for (int i = tid; i < WH / 8; i += blockDim.x) ((f4*)Wb)[i] = ((const f4*)Wsw)[i];
  for (int i = tid; i < 128; i += blockDim.x) {
    pL[i] = b0[i]; pL[128 + i] = b1[i]; pL[256 + i] = b2[i]; pL[384 + i] = b3[i];
    pL[512 + i] = g[i]; pL[640 + i] = be[i];
  }
  __syncthreads();
  const f4 z4 = {0.f,0.f,0.f,0.f};
  f4 gq = *(const f4*)(pL + 512 + 4 * lane);
  f4 bq = *(const f4*)(pL + 640 + 4 * lane);
  for (int tg = blockIdx.x; tg < nTiles; tg += gridDim.x) {
    int n0 = tg * 128 + wave * 16;
    for (int r = 0; r < 16; ++r) {
      int n = n0 + r; bool ok = n < N;
      f4 vn = z4, va = z4;
      if (ok) {
        vn = ((const f4*)(nf + (size_t)n * 128))[lane];
        f4* pa = (f4*)(aggr + (size_t)n * 128);
        va = pa[lane];
        pa[lane] = z4;            // rezero for next layer
      }
      h4* xr = (h4*)(X + r * XSH);
      xr[lane]      = cvt_h4(vn);
      xr[32 + lane] = cvt_h4(va);
    }
    mlp_chain<4, false>(X, XSH, Wb, 256, pL, lane);
    float* Xf = (float*)X; const int XSF = XSH >> 1;
    row_stats(Xf, XSF, stats, lane);
    __syncthreads();
    for (int r = 0; r < 16; ++r) {
      int n = n0 + r; if (n >= N) break;
      float mu = stats[r * 2], rs = stats[r * 2 + 1];
      f4 xv = *(const f4*)(Xf + r * XSF + 4 * lane);
      f4* pn = (f4*)(nf + (size_t)n * 128);
      f4 o = pn[lane];
      o.x += (xv.x - mu) * rs * gq.x + bq.x;
      o.y += (xv.y - mu) * rs * gq.y + bq.y;
      o.z += (xv.z - mu) * rs * gq.z + bq.z;
      o.w += (xv.w - mu) * rs * gq.w + bq.w;
      pn[lane] = o;
    }
    __syncthreads();
  }
}

// ---- decoder: 3 relu GEMMs (WMMA) + scalar 128->2 ----
__global__ void __launch_bounds__(256, 1)
k_decoder(const float* __restrict__ nf, float* __restrict__ out, int N, int nTiles,
          const _Float16* __restrict__ Wsw,
          const float* b0, const float* b1, const float* b2,
          const float* __restrict__ W3, const float* __restrict__ b3) {
  extern __shared__ char lds[];
  const int WH = 3 * 128 * 128;               // 49152 halves
  const int XSH = 280;
  _Float16* Wb   = (_Float16*)lds;
  _Float16* Xall = (_Float16*)(lds + WH * 2);
  float* pL      = (float*)(lds + WH * 2 + 8 * 16 * XSH * 2);
  int tid = threadIdx.x, lane = tid & 31, wave = tid >> 5;
  _Float16* X = Xall + wave * 16 * XSH;
  for (int i = tid; i < WH / 8; i += blockDim.x) ((f4*)Wb)[i] = ((const f4*)Wsw)[i];
  for (int i = tid; i < 128; i += blockDim.x) {
    pL[i] = b0[i]; pL[128 + i] = b1[i]; pL[256 + i] = b2[i]; pL[384 + i] = 0.f;
  }
  __syncthreads();
  for (int tg = blockIdx.x; tg < nTiles; tg += gridDim.x) {
    int n0 = tg * 128 + wave * 16;
    for (int r = 0; r < 16; ++r) {
      int n = n0 + r; bool ok = n < N;
      f4 vn = {0.f,0.f,0.f,0.f};
      if (ok) vn = ((const f4*)(nf + (size_t)n * 128))[lane];
      ((h4*)(X + r * XSH))[lane] = cvt_h4(vn);
    }
    mlp_chain<3, true>(X, XSH, Wb, 128, pL, lane);
    __syncthreads();
    float* Xf = (float*)X; const int XSF = XSH >> 1;
    if (lane < 16) {
      int n = n0 + lane;
      if (n < N) {
        const float* row = Xf + lane * XSF;
        float o0 = b3[0], o1 = b3[1];
        for (int c = 0; c < 128; ++c) {
          float v = row[c];
          o0 += v * W3[c * 2 + 0];
          o1 += v * W3[c * 2 + 1];
        }
        out[(size_t)n * 2 + 0] = o0;
        out[(size_t)n * 2 + 1] = o1;
      }
    }
    __syncthreads();
  }
}

// d_in pytree-leaf order (dict keys sorted: W,b,be,g):
// 0 x, 1 pos, 2 edge_index, 3 edge_attr, 4 emb,
// node_in 5-14 (W0..3,b0..3,be,g), edge_in 15-24,
// proc_edge 25-34, proc_node 35-44, decoder 45-52 (W0..3,b0..3)
extern "C" void kernel_launch(void* const* d_in, const int* in_sizes, int n_in,
                              void* d_out, int out_size, void* d_ws, size_t ws_size,
                              hipStream_t stream) {
  (void)n_in; (void)out_size; (void)ws_size;
  const int N = in_sizes[0];
  const int E = in_sizes[3] / 3;
  const int* x = (const int*)d_in[0];
  const float* pos = (const float*)d_in[1];
  const int* eidx = (const int*)d_in[2];
  const float* eattr = (const float*)d_in[3];
  const float* emb = (const float*)d_in[4];

  float* nf   = (float*)d_ws;
  float* aggr = nf + (size_t)N * 128;
  float* ef   = aggr + (size_t)N * 128;
  _Float16* swz = (_Float16*)(ef + (size_t)E * 128);

  _Float16* sw_node = swz;                    // 32*128 + 3*16384 = 53248
  _Float16* sw_edge = sw_node + 53248;        // 53248
  _Float16* sw_pe   = sw_edge + 53248;        // 10 * 98304
  _Float16* sw_pn   = sw_pe + 983040;         // 10 * 81920
  _Float16* sw_dec  = sw_pn + 819200;         // 49152

  auto F = [&](int i) { return (const float*)d_in[i]; };
  auto SW = [&](int i, _Float16* dst, int K, int Kpad, int batch, long long dstStride) {
    long long total = (long long)batch * Kpad * 128;
    int blocks = (int)((total + 255) / 256);
    if (blocks > 1024) blocks = 1024;
    k_swizzle<<<blocks, 256, 0, stream>>>(F(i), dst, K, Kpad, batch, dstStride);
  };

  SW(5,  sw_node,                 22, 32,  1, 0);
  SW(6,  sw_node + 4096,         128, 128, 1, 0);
  SW(7,  sw_node + 4096 + 16384, 128, 128, 1, 0);
  SW(8,  sw_node + 4096 + 32768, 128, 128, 1, 0);
  SW(15, sw_edge,                  3, 32,  1, 0);
  SW(16, sw_edge + 4096,         128, 128, 1, 0);
  SW(17, sw_edge + 4096 + 16384, 128, 128, 1, 0);
  SW(18, sw_edge + 4096 + 32768, 128, 128, 1, 0);
  SW(25, sw_pe,                  384, 384, 10, 98304);
  SW(26, sw_pe + 49152,          128, 128, 10, 98304);
  SW(27, sw_pe + 49152 + 16384,  128, 128, 10, 98304);
  SW(28, sw_pe + 49152 + 32768,  128, 128, 10, 98304);
  SW(35, sw_pn,                  256, 256, 10, 81920);
  SW(36, sw_pn + 32768,          128, 128, 10, 81920);
  SW(37, sw_pn + 32768 + 16384,  128, 128, 10, 81920);
  SW(38, sw_pn + 32768 + 32768,  128, 128, 10, 81920);
  SW(45, sw_dec,                 128, 128, 1, 0);
  SW(46, sw_dec + 16384,         128, 128, 1, 0);
  SW(47, sw_dec + 32768,         128, 128, 1, 0);

  k_zero<<<1024, 256, 0, stream>>>(aggr, (long long)N * 128);

  const int ENC_LDS  = 53248 * 2 + 8 * 16 * 280 * 2 + 8 * 32 * 4 + 768 * 4; // 182272 B
  const int EDGE_LDS = 98304 * 2 + 8 * 16 * 408 * 2 + 8 * 32 * 4 + 768 * 4; // 305152 B
  const int NODE_LDS = 81920 * 2 + 8 * 16 * 280 * 2 + 8 * 32 * 4 + 768 * 4; // 239616 B
  const int DEC_LDS  = 49152 * 2 + 8 * 16 * 280 * 2 + 768 * 4;              // 173056 B

  int nTilesN = (N + 127) / 128;
  k_node_enc<<<nTilesN, 256, ENC_LDS, stream>>>(x, pos, emb, nf, N, nTilesN,
      sw_node, F(9), F(10), F(11), F(12), F(13), F(14));

  int nTilesEt = (E + 127) / 128;
  int gE = nTilesEt < 1024 ? nTilesEt : 1024;
  k_edge_enc<<<gE, 256, ENC_LDS, stream>>>(eattr, ef, E, nTilesEt,
      sw_edge, F(19), F(20), F(21), F(22), F(23), F(24));

  for (int l = 0; l < 10; ++l) {
    k_edge_layer<<<gE, 256, EDGE_LDS, stream>>>(nf, ef, aggr, eidx, E, nTilesEt,
        sw_pe + (size_t)l * 98304,
        F(29) + l * 128, F(30) + l * 128, F(31) + l * 128, F(32) + l * 128,
        F(33) + l * 128, F(34) + l * 128);
    k_node_layer<<<nTilesN, 256, NODE_LDS, stream>>>(nf, aggr, N, nTilesN,
        sw_pn + (size_t)l * 81920,
        F(39) + l * 128, F(40) + l * 128, F(41) + l * 128, F(42) + l * 128,
        F(43) + l * 128, F(44) + l * 128);
  }

  k_decoder<<<nTilesN, 256, DEC_LDS, stream>>>(nf, (float*)d_out, N, nTilesN,
      sw_dec, F(49), F(50), F(51), F(48), F(52));
}
